// MultiQueryAttention_8323646620394
// MI455X (gfx1250) — matrix operand order repaired and checked by reading, not compiled, not run on hardware
//
#include <hip/hip_runtime.h>

// ---------------------------------------------------------------------------
// Multi-query attention on gfx1250 via bf16 WMMA (v_wmma_f32_16x16x32_bf16),
// with CDNA5 async global->LDS staging (GLOBAL_LOAD_ASYNC_TO_LDS_B128,
// ASYNCcnt) for all shared matrix tiles.
// Pipeline: f32->bf16 convert, Q/K/V projections (WMMA GEMM), flash-attention
// (WMMA scores + online softmax + WMMA PV), output projection (WMMA GEMM, f32).
// ---------------------------------------------------------------------------

typedef unsigned short u16t;
typedef __attribute__((ext_vector_type(16))) __bf16        v16bf;
typedef __attribute__((ext_vector_type(16))) unsigned short v16us;
typedef __attribute__((ext_vector_type(8)))  float          v8f;

#define EMBED 1024
#define HEADS 16
#define HDIM  64
#define BATCH 2
#define SEQ   2048
#define ROWS  (BATCH * SEQ)   // 4096

// ---- bf16 helpers (round-to-nearest-even) ----------------------------------
__device__ __forceinline__ u16t f2bf(float f) {
    unsigned int u = __float_as_uint(f);
    u += 0x7FFFu + ((u >> 16) & 1u);
    return (u16t)(u >> 16);
}

__device__ __forceinline__ v16bf bc(v16us u) {
    return __builtin_bit_cast(v16bf, u);
}

// 16 contiguous bf16 (32 bytes, 32B-aligned)
__device__ __forceinline__ v16us load16c(const u16t* p) {
    return *(const v16us*)p;
}

// two contiguous 8-element groups (A-fragment halves)
__device__ __forceinline__ v16us load8x2(const u16t* p0, const u16t* p1) {
    v16us r;
#pragma unroll
    for (int i = 0; i < 8; ++i) { r[i] = p0[i]; r[8 + i] = p1[i]; }
    return r;
}

__device__ __forceinline__ v8f wmma_bf16(v16bf a, v16bf b, v8f c) {
    // D = A(16x32) * B(32x16) + C, f32 accumulate
    return __builtin_amdgcn_wmma_f32_16x16x32_bf16(
        false, a, false, b, (short)0, c, false, false);
}

// ---- CDNA5 async global->LDS copy (ASYNCcnt path) ---------------------------
// LDS byte offset = low 32 bits of the generic shared-aperture address
__device__ __forceinline__ unsigned lds_off(const void* p) {
    return (unsigned)(unsigned long long)p;
}

__device__ __forceinline__ void async_load_b128(unsigned dst_lds,
                                                const void* src_global) {
    asm volatile("global_load_async_to_lds_b128 %0, %1, off"
                 :: "v"(dst_lds), "v"((unsigned long long)src_global)
                 : "memory");
}

__device__ __forceinline__ void wait_async0() {
    asm volatile("s_wait_asynccnt 0" ::: "memory");
}

// ---- f32 -> bf16 conversion -------------------------------------------------
__global__ void cvt_f32_bf16(const float* __restrict__ src,
                             u16t* __restrict__ dst, int n) {
    int i = blockIdx.x * blockDim.x + threadIdx.x;
    if (i < n) dst[i] = f2bf(src[i]);
}

// ---- generic WMMA GEMM: C[M,N] = A[M,K] * B[K,N] + bias[N] ------------------
// A,B bf16 row-major. Block = 4 waves; the block's 32x64 B tile is staged in
// LDS via async copy and shared by all waves; each wave owns one 16x16 tile.
// Requires: N % 64 == 0, M % 16 == 0, K % 32 == 0. grid = (M/16, N/64).
template <bool OUT_F32>
__global__ void gemm_bf16(const u16t* __restrict__ A, const u16t* __restrict__ B,
                          const float* __restrict__ bias, void* __restrict__ Cout,
                          int M, int N, int Kd) {
    __shared__ u16t Bt[32 * 64];   // 4 KB: B[kk..kk+32) x [nblk..nblk+64)

    const int tid  = threadIdx.x;
    const int lane = tid & 31;
    const int wave = tid >> 5;
    const int m0   = blockIdx.x * 16;
    const int nblk = blockIdx.y * 64;

    const int mrow = m0 + (lane & 15);
    const int nl   = wave * 16 + (lane & 15);   // column within block tile
    const int ncol = nblk + nl;
    const int kb8  = (lane < 16) ? 0 : 8;       // unused; A-frag also uses kb16
    const int kb16 = (lane < 16) ? 0 : 16;      // B-frag K base within chunk

    // cooperative staging map: 128 threads x 16B, 2 passes cover 32x64 bf16
    const int srow = tid >> 3;                  // 0..15
    const int scol = (tid & 7) * 8;             // element offset in row
    const unsigned bt0 = lds_off(Bt);

    v8f c = {0.f, 0.f, 0.f, 0.f, 0.f, 0.f, 0.f, 0.f};

    for (int kk = 0; kk < Kd; kk += 32) {
        __syncthreads();   // previous tile's readers done
#pragma unroll
        for (int p = 0; p < 2; ++p) {
            const int row = p * 16 + srow;
            async_load_b128(bt0 + (unsigned)(row * 64 + scol) * 2u,
                            B + (size_t)(kk + row) * N + nblk + scol);
        }
        wait_async0();
        __syncthreads();   // all waves' tiles visible

        const u16t* ap = A + (size_t)mrow * Kd + kk;
        v16us au = load16c(ap + kb16);
        v16us bu;
#pragma unroll
        for (int e = 0; e < 16; ++e)
            bu[e] = Bt[(kb16 + e) * 64 + nl];
        c = wmma_bf16(bc(au), bc(bu), c);
    }

    const float bv = bias[ncol];
#pragma unroll
    for (int r = 0; r < 8; ++r) {
        const int m = m0 + r + ((lane < 16) ? 0 : 8);
        const float val = c[r] + bv;
        if (OUT_F32)
            ((float*)Cout)[(size_t)m * N + ncol] = val;
        else
            ((u16t*)Cout)[(size_t)m * N + ncol] = f2bf(val);
    }
}

// ---- flash attention: 4 waves/block, one 16-query tile each, same (b,h) -----
// Q: [B*S, EMBED] bf16 (head h at column h*64). K,V: [B*S, 64] bf16 (shared
// single head). O: [B*S, EMBED] bf16. K/V tiles (32x64) are async-staged into
// LDS once per block and shared by all 4 waves.
__global__ void mqa_attention(const u16t* __restrict__ Q, const u16t* __restrict__ Km,
                              const u16t* __restrict__ Vm, u16t* __restrict__ O) {
    __shared__ u16t Kt[32 * 64];         // 4 KB key tile (row-major, d contiguous)
    __shared__ u16t Vt[32 * 64];         // 4 KB value tile
    __shared__ u16t ptile[4][16 * 32];   // per-wave P tile (bf16)

    const int tid  = threadIdx.x;
    const int lane = tid & 31;
    const int wave = tid >> 5;
    const int h = blockIdx.y;
    const int b = blockIdx.z;
    const int q0 = (blockIdx.x * 4 + wave) * 16;

    const int keyn  = lane & 15;
    const int kb8   = (lane < 16) ? 0 : 8;
    const int kb16  = (lane < 16) ? 0 : 16;
    const int kside = (lane < 16) ? 0 : 16;   // B-frag d offset for score tiles

    const u16t* Kbase = Km + (size_t)b * SEQ * HDIM;
    const u16t* Vbase = Vm + (size_t)b * SEQ * HDIM;

    // staging map: 128 threads x 16B, 2 passes per 32x64 tile
    const int srow = tid >> 3;
    const int scol = (tid & 7) * 8;
    const unsigned kt0 = lds_off(Kt);
    const unsigned vt0 = lds_off(Vt);

    // Loop-invariant Q A-fragments: d-chunks [0,32) and [32,64)
    const int mq = q0 + (lane & 15);
    const u16t* qp = Q + ((size_t)(b * SEQ) + mq) * EMBED + (size_t)h * HDIM;
    v16bf aq[2];
#pragma unroll
    for (int dc = 0; dc < 2; ++dc) {
        const u16t* base = qp + dc * 32;
        aq[dc] = bc(load16c(base + kb16));
    }

    v8f acc[4];
#pragma unroll
    for (int c = 0; c < 4; ++c) acc[c] = (v8f){0.f,0.f,0.f,0.f,0.f,0.f,0.f,0.f};
    float mrun[8], lrun[8];
#pragma unroll
    for (int r = 0; r < 8; ++r) { mrun[r] = -1e30f; lrun[r] = 0.f; }

    const float scale = 0.125f;  // 1/sqrt(64)

    for (int kt = 0; kt < SEQ; kt += 32) {
        // ---- async-stage K and V 32x64 tiles into LDS ----------------------
        __syncthreads();   // previous iteration's tile readers done
#pragma unroll
        for (int p = 0; p < 2; ++p) {
            const int row = p * 16 + srow;
            const size_t goff = (size_t)(kt + row) * HDIM + scol;
            const unsigned loff = (unsigned)(row * HDIM + scol) * 2u;
            async_load_b128(kt0 + loff, Kbase + goff);
            async_load_b128(vt0 + loff, Vbase + goff);
        }
        wait_async0();
        __syncthreads();   // all waves' tiles visible

        // ---- scores S[16 x 32] = Q(16x64) . K^T(64x32), two N-halves -------
        v8f s[2];
#pragma unroll
        for (int nh = 0; nh < 2; ++nh) {
            s[nh] = (v8f){0.f,0.f,0.f,0.f,0.f,0.f,0.f,0.f};
            const u16t* kp = Kt + (nh * 16 + keyn) * HDIM + kside;
#pragma unroll
            for (int dc = 0; dc < 2; ++dc) {
                v16bf bk = bc(load16c(kp + dc * 32));
                s[nh] = wmma_bf16(aq[dc], bk, s[nh]);
            }
        }

        // ---- online softmax (row reductions stay within lane halves) -------
        float ef[8];
#pragma unroll
        for (int r = 0; r < 8; ++r) {
            s[0][r] *= scale;
            s[1][r] *= scale;
            float v = fmaxf(s[0][r], s[1][r]);
#pragma unroll
            for (int off = 1; off < 16; off <<= 1)
                v = fmaxf(v, __shfl_xor(v, off, 32));
            const float mn = fmaxf(mrun[r], v);
            ef[r] = __expf(mrun[r] - mn);
            mrun[r] = mn;
            s[0][r] = __expf(s[0][r] - mn);
            s[1][r] = __expf(s[1][r] - mn);
            float t = s[0][r] + s[1][r];
#pragma unroll
            for (int off = 1; off < 16; off <<= 1)
                t += __shfl_xor(t, off, 32);
            lrun[r] = lrun[r] * ef[r] + t;
        }
#pragma unroll
        for (int c = 0; c < 4; ++c)
#pragma unroll
            for (int r = 0; r < 8; ++r) acc[c][r] *= ef[r];

        // ---- re-fragment P via LDS: C-layout -> A-layout (16x32 bf16) ------
        u16t* pt = ptile[wave];
#pragma unroll
        for (int r = 0; r < 8; ++r) {
            const int m = r + ((lane < 16) ? 0 : 8);
            pt[m * 32 + keyn]      = f2bf(s[0][r]);
            pt[m * 32 + 16 + keyn] = f2bf(s[1][r]);
        }
        __syncthreads();
        const u16t* pl = pt + (size_t)(lane & 15) * 32;
        v16bf ap = bc(load16c(pl + kb16));

        // ---- acc += P(16x32) . V(32x64), four d-chunks of 16 ---------------
#pragma unroll
        for (int nc = 0; nc < 4; ++nc) {
            const int d = nc * 16 + keyn;
            v16us vu;
#pragma unroll
            for (int e = 0; e < 16; ++e)
                vu[e] = Vt[(kb16 + e) * HDIM + d];
            acc[nc] = wmma_bf16(ap, bc(vu), acc[nc]);
        }
    }

    // ---- normalize and store O[b,q, h*64 + d] as bf16 ----------------------
#pragma unroll
    for (int c = 0; c < 4; ++c) {
#pragma unroll
        for (int r = 0; r < 8; ++r) {
            const int m = r + ((lane < 16) ? 0 : 8);
            const int q = q0 + m;
            const int d = c * 16 + keyn;
            const float val = acc[c][r] / lrun[r];
            O[((size_t)(b * SEQ) + q) * EMBED + (size_t)h * HDIM + d] = f2bf(val);
        }
    }
}

// ---------------------------------------------------------------------------
extern "C" void kernel_launch(void* const* d_in, const int* in_sizes, int n_in,
                              void* d_out, int out_size, void* d_ws, size_t ws_size,
                              hipStream_t stream) {
    const float* x  = (const float*)d_in[0];
    const float* Wq = (const float*)d_in[1];
    const float* bq = (const float*)d_in[2];
    const float* Wk = (const float*)d_in[3];
    const float* bk = (const float*)d_in[4];
    const float* Wv = (const float*)d_in[5];
    const float* bv = (const float*)d_in[6];
    const float* Wo = (const float*)d_in[7];
    const float* bo = (const float*)d_in[8];
    float* out = (float*)d_out;

    // workspace layout (bf16 buffers), all offsets 256B-aligned
    char* ws = (char*)d_ws;
    size_t off = 0;
    u16t* xb   = (u16t*)(ws + off); off += (size_t)ROWS * EMBED * 2;   // 8 MB
    u16t* Wqb  = (u16t*)(ws + off); off += (size_t)EMBED * EMBED * 2;  // 2 MB
    u16t* Wkb  = (u16t*)(ws + off); off += (size_t)EMBED * HDIM * 2;
    u16t* Wvb  = (u16t*)(ws + off); off += (size_t)EMBED * HDIM * 2;
    u16t* Wob  = (u16t*)(ws + off); off += (size_t)EMBED * EMBED * 2;  // 2 MB
    u16t* Qb   = (u16t*)(ws + off); off += (size_t)ROWS * EMBED * 2;   // 8 MB
    u16t* Kb   = (u16t*)(ws + off); off += (size_t)ROWS * HDIM * 2;
    u16t* Vb   = (u16t*)(ws + off); off += (size_t)ROWS * HDIM * 2;
    u16t* AO   = (u16t*)(ws + off); off += (size_t)ROWS * EMBED * 2;   // 8 MB

    // 1) convert inputs/weights to bf16
    {
        int n;
        n = ROWS * EMBED;  cvt_f32_bf16<<<(n + 255) / 256, 256, 0, stream>>>(x,  xb,  n);
        n = EMBED * EMBED; cvt_f32_bf16<<<(n + 255) / 256, 256, 0, stream>>>(Wq, Wqb, n);
        n = EMBED * HDIM;  cvt_f32_bf16<<<(n + 255) / 256, 256, 0, stream>>>(Wk, Wkb, n);
        n = EMBED * HDIM;  cvt_f32_bf16<<<(n + 255) / 256, 256, 0, stream>>>(Wv, Wvb, n);
        n = EMBED * EMBED; cvt_f32_bf16<<<(n + 255) / 256, 256, 0, stream>>>(Wo, Wob, n);
    }

    // 2) projections: Q = x*Wq+bq (bf16), K = x*Wk+bk, V = x*Wv+bv
    gemm_bf16<false><<<dim3(ROWS / 16, EMBED / 64), 128, 0, stream>>>(
        xb, Wqb, bq, Qb, ROWS, EMBED, EMBED);
    gemm_bf16<false><<<dim3(ROWS / 16, 1), 128, 0, stream>>>(
        xb, Wkb, bk, Kb, ROWS, HDIM, EMBED);
    gemm_bf16<false><<<dim3(ROWS / 16, 1), 128, 0, stream>>>(
        xb, Wvb, bv, Vb, ROWS, HDIM, EMBED);

    // 3) flash attention (shared single K/V head across 16 query heads)
    mqa_attention<<<dim3(SEQ / 64, HEADS, BATCH), 128, 0, stream>>>(Qb, Kb, Vb, AO);

    // 4) output projection: out = AO*Wo + bo (f32)
    gemm_bf16<true><<<dim3(ROWS / 16, EMBED / 64), 128, 0, stream>>>(
        AO, Wob, bo, out, ROWS, EMBED, EMBED);
}
